// Conv2d_mvm_47811575939412
// MI455X (gfx1250) — 8-bit matrix operand order repaired and checked by reading, compiled, not run on hardware
//
#include <hip/hip_runtime.h>
#include <stdint.h>


typedef __attribute__((ext_vector_type(8))) int v8i;

#define KTOT    288
#define KTILES  5
#define MTILES  4

#define PROWS    35
#define PSTRIDE  (PROWS * 34 * 32)
#define PLANE_BYTES (8 * PSTRIDE)

#define AFRAG_WORDS (MTILES * KTILES * 32 * 8)

#define XLO_OFF  0
#define XHI_OFF  (PLANE_BYTES)
#define ALO_OFF  (2 * PLANE_BYTES)
#define AHI_OFF  (2 * PLANE_BYTES + AFRAG_WORDS*4)

__device__ __forceinline__ int quant16(float v) {
    float q = rintf(v * 256.0f);
    q = fminf(fmaxf(q, -32768.0f), 32767.0f);
    return (int)q;
}

__global__ void prep_x_kernel(const float* __restrict__ x,
                              uint8_t* __restrict__ xlo,
                              uint8_t* __restrict__ xhi) {
    int idx = blockIdx.x * blockDim.x + threadIdx.x;
    if (idx >= PLANE_BYTES) return;
    int c = idx & 31;
    int w = (idx >> 5) % 34;
    int h = (idx / (34 * 32)) % PROWS;
    int b = idx / PSTRIDE;
    int v = 0;
    if (h >= 1 && h <= 32 && w >= 1 && w <= 32) {
        v = quant16(x[((b * 32 + c) * 32 + (h - 1)) * 32 + (w - 1)]);
    }
    xlo[idx] = (uint8_t)((unsigned)v & 0xFFu);
    xhi[idx] = (uint8_t)(((unsigned)v >> 8) & 0xFFu);
}

__global__ void prep_w_kernel(const float* __restrict__ weight,
                              uint32_t* __restrict__ A_lo,
                              uint32_t* __restrict__ A_hi) {
    int idx = blockIdx.x * blockDim.x + threadIdx.x;
    if (idx >= AFRAG_WORDS) return;
    int v    = idx & 7;
    int lane = (idx >> 3) & 31;
    int t    = (idx >> 8) % KTILES;
    int m    = idx / (KTILES * 32 * 8);
    int hiL  = lane >> 4;
    int M    = m * 16 + (lane & 15);
    int kbase = t * 64 + ((v >> 2) << 5) + (hiL << 4) + ((v & 3) << 2);
    uint32_t lo = 0, hi = 0;
#pragma unroll
    for (int j = 0; j < 4; ++j) {
        int K = kbase + j;
        int wq = 0;
        if (K < KTOT) {
            int r  = K >> 5;
            int c  = K & 31;
            int iy = r / 3;
            int ix = r - 3 * iy;
            wq = quant16(weight[((M * 32 + c) * 3 + iy) * 3 + ix]);
        }
        lo |= ((uint32_t)wq & 0xFFu) << (8 * j);
        hi |= (((uint32_t)wq >> 8) & 0xFFu) << (8 * j);
    }
    A_lo[idx] = lo;
    A_hi[idx] = hi;
}

__global__ void __launch_bounds__(256)
conv_wmma_kernel(const uint8_t* __restrict__ xlo,
                 const uint8_t* __restrict__ xhi,
                 const uint32_t* __restrict__ A_lo,
                 const uint32_t* __restrict__ A_hi,
                 const float* __restrict__ bias,
                 float* __restrict__ out) {
    const int lane = threadIdx.x & 31;
    const int wave = threadIdx.x >> 5;
    const int b    = blockIdx.x >> 3;
    const int p0   = ((blockIdx.x & 7) << 7) + (wave << 4);
    const int y    = p0 >> 5;
    const int x0   = p0 & 31;
    const int n    = lane & 15;
    const int hiL  = lane >> 4;
    const int px   = x0 + n;

    v8i acc_uu[MTILES] = {};
    v8i acc_mid[MTILES] = {};
    v8i acc_ss[MTILES] = {};

    const size_t bbase = (size_t)b * PSTRIDE + ((size_t)hiL << 4);

#pragma unroll
    for (int t = 0; t < KTILES; ++t) {
        const int r0 = 2 * t;
        const int r1 = 2 * t + 1;
        const int i0 = r0 / 3, jj0 = r0 - 3 * i0;
        const int i1 = r1 / 3, jj1 = r1 - 3 * i1;
        const size_t off0 = bbase + (((size_t)(y + i0) * 34 + (px + jj0)) << 5);
        const size_t off1 = bbase + (((size_t)(y + i1) * 34 + (px + jj1)) << 5);

        uint4 bl0 = *(const uint4*)(xlo + off0);
        uint4 bl1 = *(const uint4*)(xlo + off1);
        uint4 bh0 = *(const uint4*)(xhi + off0);
        uint4 bh1 = *(const uint4*)(xhi + off1);

        v8i Blo, Bhi;
        Blo[0] = (int)bl0.x; Blo[1] = (int)bl0.y; Blo[2] = (int)bl0.z; Blo[3] = (int)bl0.w;
        Blo[4] = (int)bl1.x; Blo[5] = (int)bl1.y; Blo[6] = (int)bl1.z; Blo[7] = (int)bl1.w;
        Bhi[0] = (int)bh0.x; Bhi[1] = (int)bh0.y; Bhi[2] = (int)bh0.z; Bhi[3] = (int)bh0.w;
        Bhi[4] = (int)bh1.x; Bhi[5] = (int)bh1.y; Bhi[6] = (int)bh1.z; Bhi[7] = (int)bh1.w;

#pragma unroll
        for (int m = 0; m < MTILES; ++m) {
            const uint32_t* pl = A_lo + (((m * KTILES + t) * 32 + lane) << 3);
            const uint32_t* ph = A_hi + (((m * KTILES + t) * 32 + lane) << 3);
            uint4 a0 = *(const uint4*)(pl);
            uint4 a1 = *(const uint4*)(pl + 4);
            uint4 h0 = *(const uint4*)(ph);
            uint4 h1 = *(const uint4*)(ph + 4);
            v8i Alo, Ahi;
            Alo[0] = (int)a0.x; Alo[1] = (int)a0.y; Alo[2] = (int)a0.z; Alo[3] = (int)a0.w;
            Alo[4] = (int)a1.x; Alo[5] = (int)a1.y; Alo[6] = (int)a1.z; Alo[7] = (int)a1.w;
            Ahi[0] = (int)h0.x; Ahi[1] = (int)h0.y; Ahi[2] = (int)h0.z; Ahi[3] = (int)h0.w;
            Ahi[4] = (int)h1.x; Ahi[5] = (int)h1.y; Ahi[6] = (int)h1.z; Ahi[7] = (int)h1.w;

            acc_uu[m]  = __builtin_amdgcn_wmma_i32_16x16x64_iu8(false, Alo, false, Blo,
                                                               acc_uu[m], false, false);
            acc_mid[m] = __builtin_amdgcn_wmma_i32_16x16x64_iu8(true,  Ahi, false, Blo,
                                                               acc_mid[m], false, false);
            acc_mid[m] = __builtin_amdgcn_wmma_i32_16x16x64_iu8(false, Alo, true,  Bhi,
                                                               acc_mid[m], false, false);
            acc_ss[m]  = __builtin_amdgcn_wmma_i32_16x16x64_iu8(true,  Ahi, true,  Bhi,
                                                               acc_ss[m], false, false);
        }
    }

#pragma unroll
    for (int m = 0; m < MTILES; ++m) {
#pragma unroll
        for (int j = 0; j < 8; ++j) {
            int cout = m * 16 + hiL * 8 + j;
            uint32_t s = (uint32_t)acc_uu[m][j]
                       + ((uint32_t)acc_mid[m][j] << 8)
                       + ((uint32_t)acc_ss[m][j] << 16);
            int oi = (int)s;
            int sa = oi >> 4;
            sa = sa < -32768 ? -32768 : (sa > 32767 ? 32767 : sa);
            float f = (float)sa * (1.0f / 4096.0f) + bias[cout];
            out[(((size_t)b * 64 + cout) * 32 + y) * 32 + px] = f;
        }
    }
}

extern "C" void kernel_launch(void* const* d_in, const int* in_sizes, int n_in,
                              void* d_out, int out_size, void* d_ws, size_t ws_size,
                              hipStream_t stream) {
    (void)in_sizes; (void)n_in; (void)out_size; (void)ws_size;

    const float* x      = (const float*)d_in[0];
    const float* weight = (const float*)d_in[1];
    const float* bias   = (const float*)d_in[2];
    float* out          = (float*)d_out;

    uint8_t*  xlo  = (uint8_t*)d_ws + XLO_OFF;
    uint8_t*  xhi  = (uint8_t*)d_ws + XHI_OFF;
    uint32_t* A_lo = (uint32_t*)((char*)d_ws + ALO_OFF);
    uint32_t* A_hi = (uint32_t*)((char*)d_ws + AHI_OFF);

    {
        int threads = 256;
        int blocks = (PLANE_BYTES + threads - 1) / threads;
        prep_x_kernel<<<blocks, threads, 0, stream>>>(x, xlo, xhi);
    }
    {
        int threads = 256;
        int blocks = (AFRAG_WORDS + threads - 1) / threads;
        prep_w_kernel<<<blocks, threads, 0, stream>>>(weight, A_lo, A_hi);
    }
    {
        conv_wmma_kernel<<<64, 256, 0, stream>>>(xlo, xhi, A_lo, A_hi, bias, out);
    }
}
